// TypeAwareGraphSAGE_65584150610481
// MI455X (gfx1250) — matrix operand order repaired and checked by reading, not compiled, not run on hardware
//
#include <hip/hip_runtime.h>
#include <hip/hip_bf16.h>

// ---------------- problem constants (from reference) ----------------
#define NN_NODES 50000
#define NE_EDGES 800000
#define H_DIM    128
#define C_DIM    16

typedef __attribute__((ext_vector_type(16))) __bf16 v16bf;
typedef __attribute__((ext_vector_type(8)))  __bf16 v8bf;
typedef __attribute__((ext_vector_type(8)))  float  v8f;

static __device__ __forceinline__ unsigned short f32_to_bf16_bits(float f) {
  union { float f; unsigned int u; } c; c.f = f;
  unsigned int r = c.u + 0x7FFFu + ((c.u >> 16) & 1u);   // round-to-nearest-even
  return (unsigned short)(r >> 16);
}

// ---------------- elementwise / graph kernels ----------------
__global__ void cvt_bf16_kernel(const float* __restrict__ in,
                                unsigned short* __restrict__ out, long long n) {
  long long i  = (long long)blockIdx.x * blockDim.x + threadIdx.x;
  long long st = (long long)gridDim.x * blockDim.x;
  for (; i < n; i += st) out[i] = f32_to_bf16_bits(in[i]);
}

__global__ void degree_kernel(const int* __restrict__ dst, float* __restrict__ deg, int E) {
  int i = blockIdx.x * blockDim.x + threadIdx.x;
  if (i < E) atomicAdd(&deg[dst[i]], 1.0f);
}

__global__ void invdeg_kernel(const float* __restrict__ deg, float* __restrict__ inv, int n) {
  int i = blockIdx.x * blockDim.x + threadIdx.x;
  if (i < n) inv[i] = 1.0f / fmaxf(deg[i], 1.0f);
}

// out[dst[e]*F + f] += h[src[e]*F + f]  (F = 1<<flog2), L2-resident atomics
__global__ void scatter_add_kernel(const float* __restrict__ h, const int* __restrict__ src,
                                   const int* __restrict__ dst, float* __restrict__ out,
                                   int flog2, long long total) {
  long long i  = (long long)blockIdx.x * blockDim.x + threadIdx.x;
  long long st = (long long)gridDim.x * blockDim.x;
  const long long fmask = (1LL << flog2) - 1;
  for (; i < total; i += st) {
    int e = (int)(i >> flog2);
    int f = (int)(i & fmask);
    atomicAdd(&out[((size_t)dst[e] << flog2) + f], h[((size_t)src[e] << flog2) + f]);
  }
}

// neigh_bf16[i] = bf16( neigh[i] * inv_deg[i >> flog2] )
__global__ void scale_cvt_kernel(const float* __restrict__ x, const float* __restrict__ invd,
                                 unsigned short* __restrict__ out, int flog2, long long total) {
  long long i  = (long long)blockIdx.x * blockDim.x + threadIdx.x;
  long long st = (long long)gridDim.x * blockDim.x;
  for (; i < total; i += st) {
    long long row = i >> flog2;
    out[i] = f32_to_bf16_bits(x[i] * invd[row]);
  }
}

// ---------------- bf16 WMMA GEMM: D = A[M,K] @ B[K,N] (+Cin)(+bias)(BN)(ReLU) ----------------
// wave32: each wave owns one 16x16 tile; 8 waves per block -> 128(M) x 16(N) slab.
// B panel [K,16] staged transposed in LDS -> contiguous ds_load_b128 fragments.
__global__ __launch_bounds__(256) void gemm_bf16_wmma(
    const unsigned short* __restrict__ A, const unsigned short* __restrict__ B,
    const float* __restrict__ Cin,  const float* __restrict__ bias,
    const float* __restrict__ bng,  const float* __restrict__ bnb,
    const float* __restrict__ bnm,  const float* __restrict__ bnv,
    float* __restrict__ Df, unsigned short* __restrict__ Dh,
    int M, int Nn, int K, int relu)
{
  extern __shared__ unsigned short sB[];   // transposed panel: sB[n*K + k], n in 0..15
  const int n0 = blockIdx.x * 16;
  for (int i = threadIdx.x; i < (K << 4); i += 256) {
    int kk = i >> 4, nn = i & 15;
    sB[nn * K + kk] = B[(size_t)kk * Nn + (n0 + nn)];
  }
  __syncthreads();

  const int wave = threadIdx.x >> 5;
  const int lane = threadIdx.x & 31;
  const int half = lane >> 4;      // 0: lanes 0-15, 1: lanes 16-31
  const int r    = lane & 15;
  const int m0   = (blockIdx.y * 8 + wave) * 16;
  if (m0 >= M) return;

  // A 16-bit layout: lanes 0-15 hold K[0..7]+K[16..23]; lanes 16-31 hold K[8..15]+K[24..31]
  const unsigned short* Arow = A  + (size_t)(m0 + r) * K + 8 * half;
  // B 16-bit layout: lanes 0-15 hold K[0..7]+K[16..23]; lanes 16-31 hold K[8..15]+K[24..31], col = r
  const unsigned short* Bcol = sB + (size_t)r * K + 8 * half;

  v8f acc = {};
#pragma unroll 2
  for (int k0 = 0; k0 < K; k0 += 32) {
    v8bf a0 = *(const v8bf*)(Arow + k0);
    v8bf a1 = *(const v8bf*)(Arow + k0 + 16);
    v8bf b0 = *(const v8bf*)(Bcol + k0);
    v8bf b1 = *(const v8bf*)(Bcol + k0 + 16);
    v16bf a, b;
#pragma unroll
    for (int e = 0; e < 8; ++e) {
      a[e] = a0[e]; a[e + 8] = a1[e];
      b[e] = b0[e]; b[e + 8] = b1[e];
    }
    acc = __builtin_amdgcn_wmma_f32_16x16x32_bf16(false, a, false, b,
                                                  (short)0, acc, false, false);
  }

  // epilogue: per-column BN folded to scale/shift
  const int col = n0 + r;
  float bsc = 1.0f, bsh = 0.0f;
  if (bng) {
    float iv = rsqrtf(bnv[col] + 1e-5f);
    bsc = bng[col] * iv;
    bsh = bnb[col] - bnm[col] * bsc;
  }
  const float bb = bias ? bias[col] : 0.0f;
#pragma unroll
  for (int v = 0; v < 8; ++v) {
    size_t idx = (size_t)(m0 + v + 8 * half) * Nn + col;   // C/D layout: vgpr v -> row v+8*half
    float x = acc[v] + bb;
    if (Cin)  x += Cin[idx];
    if (bng)  x  = x * bsc + bsh;
    if (relu) x  = fmaxf(x, 0.0f);
    if (Df) Df[idx] = x;
    if (Dh) Dh[idx] = f32_to_bf16_bits(x);
  }
}

// ---------------- host-side helpers ----------------
static inline void launch_gemm(hipStream_t s,
    const unsigned short* A, const unsigned short* B,
    const float* Cin, const float* bias,
    const float* g, const float* b, const float* m, const float* v,
    float* Df, unsigned short* Dh, int M, int Nn, int K, int relu)
{
  dim3 grid(Nn / 16, ((M + 15) / 16 + 7) / 8);
  gemm_bf16_wmma<<<grid, dim3(256), (size_t)K * 16 * sizeof(unsigned short), s>>>(
      A, B, Cin, bias, g, b, m, v, Df, Dh, M, Nn, K, relu);
}

extern "C" void kernel_launch(void* const* d_in, const int* in_sizes, int n_in,
                              void* d_out, int out_size, void* d_ws, size_t ws_size,
                              hipStream_t stream) {
  (void)in_sizes; (void)n_in; (void)out_size; (void)ws_size;
  const int N = NN_NODES, E = NE_EDGES;

  // ---- inputs (setup_inputs order) ----
  const float* features = (const float*)d_in[0];
  const int*   src      = (const int*)  d_in[1];
  const int*   dst      = (const int*)  d_in[2];
  const float* W_in = (const float*)d_in[3];  const float* b_in = (const float*)d_in[4];
  const float* Ws0  = (const float*)d_in[5];  const float* Wn0  = (const float*)d_in[6];
  const float* bs0  = (const float*)d_in[7];
  const float* g0   = (const float*)d_in[8];  const float* bt0  = (const float*)d_in[9];
  const float* m0_  = (const float*)d_in[10]; const float* v0_  = (const float*)d_in[11];
  const float* Ws1  = (const float*)d_in[12]; const float* Wn1  = (const float*)d_in[13];
  const float* bs1  = (const float*)d_in[14];
  const float* g1   = (const float*)d_in[15]; const float* bt1  = (const float*)d_in[16];
  const float* m1_  = (const float*)d_in[17]; const float* v1_  = (const float*)d_in[18];
  const float* Wr   = (const float*)d_in[19]; const float* br   = (const float*)d_in[20];
  const float* Wc1  = (const float*)d_in[21]; const float* bc1  = (const float*)d_in[22];
  const float* Wc2  = (const float*)d_in[23]; const float* bc2  = (const float*)d_in[24];

  // ---- workspace carve-out ----
  char* ws = (char*)d_ws;
  size_t off = 0;
  auto carve = [&](size_t bytes) -> char* {
    char* p = ws + off;
    off = (off + bytes + 255) & ~(size_t)255;
    return p;
  };
  unsigned short* feat_bf = (unsigned short*)carve((size_t)N * 128 * 2);
  unsigned short* w_in_bf = (unsigned short*)carve(128 * 128 * 2);
  unsigned short* ws0_bf  = (unsigned short*)carve(128 * 256 * 2);
  unsigned short* wn0_bf  = (unsigned short*)carve(128 * 256 * 2);
  unsigned short* ws1_bf  = (unsigned short*)carve(256 * 128 * 2);
  unsigned short* wn1_bf  = (unsigned short*)carve(256 * 128 * 2);
  unsigned short* wr_bf   = (unsigned short*)carve(256 * 128 * 2);
  unsigned short* wc1_bf  = (unsigned short*)carve(128 * 64 * 2);
  unsigned short* wc2_bf  = (unsigned short*)carve(64 * 16 * 2);
  float* deg   = (float*)carve((size_t)N * 4);
  float* invd  = (float*)carve((size_t)N * 4);
  float* h0_f  = (float*)carve((size_t)N * 128 * 4);
  unsigned short* h0_bf = (unsigned short*)carve((size_t)N * 128 * 2);
  float* ng0_f = (float*)carve((size_t)N * 128 * 4);
  unsigned short* ng0_bf = (unsigned short*)carve((size_t)N * 128 * 2);
  float* t0    = (float*)carve((size_t)N * 256 * 4);
  float* h1_f  = (float*)carve((size_t)N * 256 * 4);
  unsigned short* h1_bf = (unsigned short*)carve((size_t)N * 256 * 2);
  float* ng1_f = (float*)carve((size_t)N * 256 * 4);
  unsigned short* ng1_bf = (unsigned short*)carve((size_t)N * 256 * 2);
  float* t1    = (float*)carve((size_t)N * 128 * 4);
  unsigned short* h2_bf = (unsigned short*)carve((size_t)N * 128 * 2);
  float* t2    = (float*)carve((size_t)N * 128 * 4);
  unsigned short* hf_bf = (unsigned short*)carve((size_t)N * 128 * 2);
  unsigned short* hc_bf = (unsigned short*)carve((size_t)N * 64 * 2);

  const dim3 EB(4096), ET(256);

  // ---- fp32 -> bf16 conversions (weights + features) ----
  cvt_bf16_kernel<<<EB, ET, 0, stream>>>(features, feat_bf, (long long)N * 128);
  cvt_bf16_kernel<<<EB, ET, 0, stream>>>(W_in, w_in_bf, 128 * 128);
  cvt_bf16_kernel<<<EB, ET, 0, stream>>>(Ws0,  ws0_bf,  128 * 256);
  cvt_bf16_kernel<<<EB, ET, 0, stream>>>(Wn0,  wn0_bf,  128 * 256);
  cvt_bf16_kernel<<<EB, ET, 0, stream>>>(Ws1,  ws1_bf,  256 * 128);
  cvt_bf16_kernel<<<EB, ET, 0, stream>>>(Wn1,  wn1_bf,  256 * 128);
  cvt_bf16_kernel<<<EB, ET, 0, stream>>>(Wr,   wr_bf,   256 * 128);
  cvt_bf16_kernel<<<EB, ET, 0, stream>>>(Wc1,  wc1_bf,  128 * 64);
  cvt_bf16_kernel<<<EB, ET, 0, stream>>>(Wc2,  wc2_bf,  64 * 16);

  // ---- degrees ----
  hipMemsetAsync(deg, 0, (size_t)N * 4, stream);
  degree_kernel<<<(E + 255) / 256, 256, 0, stream>>>(dst, deg, E);
  invdeg_kernel<<<(N + 255) / 256, 256, 0, stream>>>(deg, invd, N);

  // ---- h0 = relu(features @ W_in + b_in) ----
  launch_gemm(stream, feat_bf, w_in_bf, nullptr, b_in,
              nullptr, nullptr, nullptr, nullptr, h0_f, h0_bf, N, 128, 128, 1);

  // ---- SAGE layer 0: neigh mean, then relu(BN(h0@Ws0 + neigh@Wn0 + bs0)) ----
  hipMemsetAsync(ng0_f, 0, (size_t)N * 128 * 4, stream);
  scatter_add_kernel<<<EB, ET, 0, stream>>>(h0_f, src, dst, ng0_f, 7, (long long)E * 128);
  scale_cvt_kernel<<<EB, ET, 0, stream>>>(ng0_f, invd, ng0_bf, 7, (long long)N * 128);
  launch_gemm(stream, h0_bf, ws0_bf, nullptr, nullptr,
              nullptr, nullptr, nullptr, nullptr, t0, nullptr, N, 256, 128, 0);
  launch_gemm(stream, ng0_bf, wn0_bf, t0, bs0,
              g0, bt0, m0_, v0_, h1_f, h1_bf, N, 256, 128, 1);

  // ---- SAGE layer 1: 2H -> H ----
  hipMemsetAsync(ng1_f, 0, (size_t)N * 256 * 4, stream);
  scatter_add_kernel<<<EB, ET, 0, stream>>>(h1_f, src, dst, ng1_f, 8, (long long)E * 256);
  scale_cvt_kernel<<<EB, ET, 0, stream>>>(ng1_f, invd, ng1_bf, 8, (long long)N * 256);
  launch_gemm(stream, h1_bf, ws1_bf, nullptr, nullptr,
              nullptr, nullptr, nullptr, nullptr, t1, nullptr, N, 128, 256, 0);
  launch_gemm(stream, ng1_bf, wn1_bf, t1, bs1,
              g1, bt1, m1_, v1_, nullptr, h2_bf, N, 128, 256, 1);

  // ---- h_final = relu([h0, h2] @ Wr + br) = relu(h0@Wr_top + h2@Wr_bot + br) ----
  launch_gemm(stream, h0_bf, wr_bf, nullptr, nullptr,
              nullptr, nullptr, nullptr, nullptr, t2, nullptr, N, 128, 128, 0);
  launch_gemm(stream, h2_bf, wr_bf + 128 * 128, t2, br,
              nullptr, nullptr, nullptr, nullptr, nullptr, hf_bf, N, 128, 128, 1);

  // ---- classifier: relu(hf @ Wc1 + bc1) @ Wc2 + bc2 ----
  launch_gemm(stream, hf_bf, wc1_bf, nullptr, bc1,
              nullptr, nullptr, nullptr, nullptr, nullptr, hc_bf, N, 64, 128, 1);
  launch_gemm(stream, hc_bf, wc2_bf, nullptr, bc2,
              nullptr, nullptr, nullptr, nullptr, (float*)d_out, nullptr, N, 16, 64, 0);
}
